// GraphConvLayer_60619168416170
// MI455X (gfx1250) — matrix operand order repaired and checked by reading, not compiled, not run on hardware
//
#include <hip/hip_runtime.h>

#define NDIM 64
#define EDIM 32
#define K1E  160   // 2*NDIM + EDIM
#define K1N  128   // 2*NDIM
#define LDAE 168   // padded stride (halves): K=160 tables / edge A tile
#define LDAN 136   // padded stride (halves): K=128 table / node A tile
#define LDH  72    // padded stride (halves): K=64 tables / hidden tile
#define NB_EDGE 1280

typedef __attribute__((ext_vector_type(16))) __bf16 bf16x16;
typedef __attribute__((ext_vector_type(8)))  __bf16 bf16x8;
typedef __attribute__((ext_vector_type(8)))  float  f32x8;

#define WMMA_BF16(A, B, C) \
  __builtin_amdgcn_wmma_f32_16x16x32_bf16(false, (A), false, (B), (short)0, (C), false, false)

union AFrag { bf16x16 v; bf16x8 h[2]; };

__device__ __forceinline__ unsigned short f2bf(float f) {
  union { float f; unsigned u; } cv; cv.f = f;
  unsigned r = cv.u + 0x7FFFu + ((cv.u >> 16) & 1u);   // round-to-nearest-even
  return (unsigned short)(r >> 16);
}

__device__ __forceinline__ void st_bf4(unsigned short* p, float4 v) {
  uint2 u;
  u.x = (unsigned)f2bf(v.x) | ((unsigned)f2bf(v.y) << 16);
  u.y = (unsigned)f2bf(v.z) | ((unsigned)f2bf(v.w) << 16);
  *(uint2*)p = u;
}

// contiguous 16-half B fragment (lane = N column, halves = K)
__device__ __forceinline__ bf16x16 ldB(const unsigned short* p) {
  AFrag b;
  b.h[0] = *(const bf16x8*)p;
  b.h[1] = *(const bf16x8*)(p + 8);
  return b.v;
}

// A fragment: two 8-half chunks at K = base and K = base+8
__device__ __forceinline__ bf16x16 ldA(const unsigned short* p) {
  AFrag a;
  a.h[0] = *(const bf16x8*)p;
  a.h[1] = *(const bf16x8*)(p + 8);
  return a.v;
}

// ---------------------------------------------------------------- zero d_out
__global__ __launch_bounds__(256) void k_zero(float4* __restrict__ p, int n4) {
  int i = blockIdx.x * blockDim.x + threadIdx.x;
  if (i < n4) p[i] = make_float4(0.f, 0.f, 0.f, 0.f);
}

// ------------------------------- convert weights to bf16, transposed [N][K]
__global__ __launch_bounds__(256) void k_prep(const float* __restrict__ W1e,
                                              const float* __restrict__ W2e,
                                              const float* __restrict__ W1n,
                                              const float* __restrict__ W2n,
                                              unsigned short* __restrict__ wts) {
  const int t = threadIdx.x;
  unsigned short* t1e = wts;                    // [64][160]
  unsigned short* t2e = wts + 64 * K1E;         // [64][64]
  unsigned short* t1n = t2e + 64 * NDIM;        // [64][128]
  unsigned short* t2n = t1n + 64 * K1N;         // [64][64]
  for (int i = t; i < 64 * K1E; i += 256)  { int n = i / K1E, k = i % K1E;  t1e[i] = f2bf(W1e[k * 64 + n]); }
  for (int i = t; i < 64 * NDIM; i += 256) { int n = i / NDIM, k = i % NDIM; t2e[i] = f2bf(W2e[k * 64 + n]); }
  for (int i = t; i < 64 * K1N; i += 256)  { int n = i / K1N, k = i % K1N;  t1n[i] = f2bf(W1n[k * 64 + n]); }
  for (int i = t; i < 64 * NDIM; i += 256) { int n = i / NDIM, k = i % NDIM; t2n[i] = f2bf(W2n[k * 64 + n]); }
}

// -------------------------------------------- pre-convert x to bf16 (gather copy)
__global__ __launch_bounds__(256) void k_prex(const float4* __restrict__ x4,
                                              uint2* __restrict__ xb4, int n4) {
  int i = blockIdx.x * blockDim.x + threadIdx.x;
  if (i < n4) {
    float4 v = x4[i];
    uint2 u;
    u.x = (unsigned)f2bf(v.x) | ((unsigned)f2bf(v.y) << 16);
    u.y = (unsigned)f2bf(v.z) | ((unsigned)f2bf(v.w) << 16);
    xb4[i] = u;
  }
}

// ------------------------------------------------------------- edge MLP + scatter
// persistent: NB_EDGE blocks x 4 waves; GEMM1 weights in LDS, GEMM2 weights in VGPRs
__global__ __launch_bounds__(128) void k_edge(const unsigned short* __restrict__ xb,
                                              const int* __restrict__ eidx,
                                              const float* __restrict__ eattr,
                                              const unsigned short* __restrict__ wts,
                                              const float* __restrict__ b1e,
                                              const float* __restrict__ b2e,
                                              float* __restrict__ agg, int E) {
  __shared__ unsigned short sW[64 * LDAE + 64 * LDH];  // 30720 B
  __shared__ unsigned short sA[4 * 16 * LDAE];         // 21504 B
  __shared__ unsigned short sH[4 * 16 * LDH];          //  9216 B
  const int wave  = threadIdx.x >> 5;
  const int lane  = threadIdx.x & 31;
  const int ncol  = lane & 15;
  const int khalf = lane >> 4;
  unsigned short* wA  = sA + wave * 16 * LDAE;
  unsigned short* wH  = sH + wave * 16 * LDH;
  unsigned short* sW2 = sW + 64 * LDAE;

  // stage weight tables into LDS with padded (conflict-free) strides
  for (int i = threadIdx.x; i < 64 * 20; i += 128) {     // t1e: 20 uint4 per row
    int n = i / 20, j = i % 20;
    *(uint4*)(sW + n * LDAE + j * 8) = *(const uint4*)(wts + n * K1E + j * 8);
  }
  for (int i = threadIdx.x; i < 64 * 8; i += 128) {      // t2e: 8 uint4 per row
    int n = i / 8, j = i % 8;
    *(uint4*)(sW2 + n * LDH + j * 8) = *(const uint4*)(wts + 64 * K1E + n * NDIM + j * 8);
  }
  __syncthreads();

  // GEMM2 weights live in registers for the whole persistent loop (loop-invariant)
  bf16x16 B2r[8];
#pragma unroll
  for (int kk = 0; kk < 2; ++kk)
#pragma unroll
    for (int nt = 0; nt < 4; ++nt)
      B2r[kk * 4 + nt] = ldB(sW2 + (nt * 16 + ncol) * LDH + kk * 32 + khalf * 16);

  float bb1[4], bb2[4];
#pragma unroll
  for (int nt = 0; nt < 4; ++nt) { bb1[nt] = b1e[nt * 16 + ncol]; bb2[nt] = b2e[nt * 16 + ncol]; }

  const int nTiles = (E + 15) / 16;
  const int stride = NB_EDGE * 4;
  const int iters  = (nTiles + stride - 1) / stride;

  for (int it = 0; it < iters; ++it) {
    const int tile    = it * stride + blockIdx.x * 4 + wave;
    const bool active = tile < nTiles;
    const int ebase   = tile * 16;

    // stage A = [xb[row] | xb[col] | bf16(edge_attr)] : 16 edges x 160 halves
    {
      const int r = lane >> 1, half = lane & 1, e = ebase + r;
      if (active && e < E) {
        const int ri = eidx[e];
        const int ci = eidx[E + e];
        const uint4*  xr = (const uint4*)(xb + (size_t)ri * NDIM) + half * 4;
        const uint4*  xc = (const uint4*)(xb + (size_t)ci * NDIM) + half * 4;
        const float4* ea = (const float4*)(eattr + (size_t)e * EDIM) + half * 4;
        unsigned short* dst = wA + r * LDAE;
        uint4* d0 = (uint4*)(dst + half * 32);
        uint4* d1 = (uint4*)(dst + 64 + half * 32);
#pragma unroll
        for (int i = 0; i < 4; ++i) d0[i] = xr[i];
#pragma unroll
        for (int i = 0; i < 4; ++i) d1[i] = xc[i];
#pragma unroll
        for (int i = 0; i < 4; ++i) st_bf4(dst + 128 + half * 16 + i * 4, ea[i]);
      }
    }
    __syncthreads();

    // GEMM1: [16x160] x [160x64], software-pipelined (next-k loads overlap WMMAs)
    f32x8 acc[4] = {};
    const unsigned short* aRow = wA + ncol * LDAE + khalf * 16;
    bf16x16 aC = ldA(aRow);
    bf16x16 bC[4];
#pragma unroll
    for (int nt = 0; nt < 4; ++nt) bC[nt] = ldB(sW + (nt * 16 + ncol) * LDAE + khalf * 16);
#pragma unroll
    for (int kk = 0; kk < K1E / 32; ++kk) {
      bf16x16 aN;
      bf16x16 bN[4];
      if (kk < K1E / 32 - 1) {
        aN = ldA(aRow + (kk + 1) * 32);
#pragma unroll
        for (int nt = 0; nt < 4; ++nt)
          bN[nt] = ldB(sW + (nt * 16 + ncol) * LDAE + (kk + 1) * 32 + khalf * 16);
      }
#pragma unroll
      for (int nt = 0; nt < 4; ++nt) acc[nt] = WMMA_BF16(aC, bC[nt], acc[nt]);
      if (kk < K1E / 32 - 1) {
        aC = aN;
#pragma unroll
        for (int nt = 0; nt < 4; ++nt) bC[nt] = bN[nt];
      }
    }
#pragma unroll
    for (int nt = 0; nt < 4; ++nt) {
#pragma unroll
      for (int r = 0; r < 8; ++r) {
        float v = acc[nt][r] + bb1[nt];
        v = v > 0.f ? v : 0.f;
        wH[(khalf * 8 + r) * LDH + nt * 16 + ncol] = f2bf(v);
      }
    }
    __syncthreads();

    // GEMM2: [16x64] x [64x64], B in registers -> only 4 A-side LDS loads
    f32x8 acc2[4] = {};
    const unsigned short* hRow = wH + ncol * LDH + khalf * 16;
    bf16x16 h0 = ldA(hRow);
    bf16x16 h1 = ldA(hRow + 32);
#pragma unroll
    for (int nt = 0; nt < 4; ++nt) acc2[nt] = WMMA_BF16(h0, B2r[nt], acc2[nt]);
#pragma unroll
    for (int nt = 0; nt < 4; ++nt) acc2[nt] = WMMA_BF16(h1, B2r[4 + nt], acc2[nt]);

    // scatter-add edge_output into agg[col] (L2-resident f32 atomics)
    int dstn[8];
#pragma unroll
    for (int r = 0; r < 8; ++r) {
      const int e = ebase + khalf * 8 + r;
      dstn[r] = (active && e < E) ? eidx[E + e] : -1;
    }
#pragma unroll
    for (int nt = 0; nt < 4; ++nt) {
      const int n = nt * 16 + ncol;
#pragma unroll
      for (int r = 0; r < 8; ++r)
        if (dstn[r] >= 0)
          unsafeAtomicAdd(agg + (size_t)dstn[r] * NDIM + n, acc2[nt][r] + bb2[nt]);
    }
    __syncthreads();
  }
}

// ------------------------------------------------------------- node MLP + residual
__global__ __launch_bounds__(128) void k_node(const unsigned short* __restrict__ xb,
                                              const float* __restrict__ x,
                                              const unsigned short* __restrict__ wts,
                                              const float* __restrict__ b1n,
                                              const float* __restrict__ b2n,
                                              float* xagg_out, int N) {
  __shared__ unsigned short sW[64 * LDAN + 64 * LDH];  // 26624 B
  __shared__ unsigned short sA[4 * 16 * LDAN];         // 17408 B
  __shared__ unsigned short sH[4 * 16 * LDH];          //  9216 B
  const int wave  = threadIdx.x >> 5;
  const int lane  = threadIdx.x & 31;
  const int ncol  = lane & 15;
  const int khalf = lane >> 4;
  const int nbase = (blockIdx.x * 4 + wave) * 16;
  unsigned short* wA  = sA + wave * 16 * LDAN;
  unsigned short* wH  = sH + wave * 16 * LDH;
  unsigned short* sW2 = sW + 64 * LDAN;
  const unsigned short* t1n = wts + 64 * K1E + 64 * NDIM;
  const unsigned short* t2n = t1n + 64 * K1N;

  for (int i = threadIdx.x; i < 64 * 16; i += 128) {     // t1n: 16 uint4 per row
    int n = i / 16, j = i % 16;
    *(uint4*)(sW + n * LDAN + j * 8) = *(const uint4*)(t1n + n * K1N + j * 8);
  }
  for (int i = threadIdx.x; i < 64 * 8; i += 128) {      // t2n: 8 uint4 per row
    int n = i / 8, j = i % 8;
    *(uint4*)(sW2 + n * LDH + j * 8) = *(const uint4*)(t2n + n * NDIM + j * 8);
  }

  // stage A = [xb[v] | bf16(agg[v])] : 16 nodes x 128 halves
  {
    const int r = lane >> 1, half = lane & 1, v = nbase + r;
    unsigned short* dst = wA + r * LDAN;
    if (v < N) {
      const uint4*  xv = (const uint4*)(xb + (size_t)v * NDIM) + half * 4;
      const float4* av = (const float4*)(xagg_out + (size_t)v * NDIM) + half * 8;
      uint4* d0 = (uint4*)(dst + half * 32);
#pragma unroll
      for (int i = 0; i < 4; ++i) d0[i] = xv[i];
#pragma unroll
      for (int i = 0; i < 8; ++i) st_bf4(dst + 64 + half * 32 + i * 4, av[i]);
    } else {
      uint2* z = (uint2*)(dst + half * 64);
#pragma unroll
      for (int i = 0; i < 16; ++i) z[i] = make_uint2(0u, 0u);
    }
  }
  __syncthreads();

  // GEMM2 weights into registers
  bf16x16 B2r[8];
#pragma unroll
  for (int kk = 0; kk < 2; ++kk)
#pragma unroll
    for (int nt = 0; nt < 4; ++nt)
      B2r[kk * 4 + nt] = ldB(sW2 + (nt * 16 + ncol) * LDH + kk * 32 + khalf * 16);

  // GEMM1: [16x128] x [128x64], software-pipelined
  f32x8 acc[4] = {};
  const unsigned short* aRow = wA + ncol * LDAN + khalf * 16;
  bf16x16 aC = ldA(aRow);
  bf16x16 bC[4];
#pragma unroll
  for (int nt = 0; nt < 4; ++nt) bC[nt] = ldB(sW + (nt * 16 + ncol) * LDAN + khalf * 16);
#pragma unroll
  for (int kk = 0; kk < K1N / 32; ++kk) {
    bf16x16 aN;
    bf16x16 bN[4];
    if (kk < K1N / 32 - 1) {
      aN = ldA(aRow + (kk + 1) * 32);
#pragma unroll
      for (int nt = 0; nt < 4; ++nt)
        bN[nt] = ldB(sW + (nt * 16 + ncol) * LDAN + (kk + 1) * 32 + khalf * 16);
    }
#pragma unroll
    for (int nt = 0; nt < 4; ++nt) acc[nt] = WMMA_BF16(aC, bC[nt], acc[nt]);
    if (kk < K1N / 32 - 1) {
      aC = aN;
#pragma unroll
      for (int nt = 0; nt < 4; ++nt) bC[nt] = bN[nt];
    }
  }
#pragma unroll
  for (int nt = 0; nt < 4; ++nt) {
    const float bb = b1n[nt * 16 + ncol];
#pragma unroll
    for (int r = 0; r < 8; ++r) {
      float v = acc[nt][r] + bb;
      v = v > 0.f ? v : 0.f;
      wH[(khalf * 8 + r) * LDH + nt * 16 + ncol] = f2bf(v);
    }
  }
  __syncthreads();

  // GEMM2: [16x64] x [64x64], then +b2n +x residual, relu, store
  f32x8 acc2[4] = {};
  const unsigned short* hRow = wH + ncol * LDH + khalf * 16;
  bf16x16 h0 = ldA(hRow);
  bf16x16 h1 = ldA(hRow + 32);
#pragma unroll
  for (int nt = 0; nt < 4; ++nt) acc2[nt] = WMMA_BF16(h0, B2r[nt], acc2[nt]);
#pragma unroll
  for (int nt = 0; nt < 4; ++nt) acc2[nt] = WMMA_BF16(h1, B2r[4 + nt], acc2[nt]);

#pragma unroll
  for (int nt = 0; nt < 4; ++nt) {
    const int n = nt * 16 + ncol;
    const float bb = b2n[n];
#pragma unroll
    for (int r = 0; r < 8; ++r) {
      const int v = nbase + khalf * 8 + r;
      if (v < N) {
        float res = acc2[nt][r] + bb + x[(size_t)v * NDIM + n];
        xagg_out[(size_t)v * NDIM + n] = res > 0.f ? res : 0.f;
      }
    }
  }
}

extern "C" void kernel_launch(void* const* d_in, const int* in_sizes, int n_in,
                              void* d_out, int out_size, void* d_ws, size_t ws_size,
                              hipStream_t stream) {
  const float* x     = (const float*)d_in[0];
  const int*   eidx  = (const int*)d_in[1];
  const float* eattr = (const float*)d_in[2];
  const float* W1e = (const float*)d_in[3];
  const float* b1e = (const float*)d_in[4];
  const float* W2e = (const float*)d_in[5];
  const float* b2e = (const float*)d_in[6];
  const float* W1n = (const float*)d_in[7];
  const float* b1n = (const float*)d_in[8];
  const float* W2n = (const float*)d_in[9];
  const float* b2n = (const float*)d_in[10];
  float* out = (float*)d_out;                    // doubles as the aggregation buffer

  unsigned short* wts = (unsigned short*)d_ws;   // bf16 weight tables (~52 KB)
  unsigned short* xb  = wts + 64 * K1E + 64 * NDIM + 64 * K1N + 64 * NDIM;  // bf16 copy of x

  const int E = in_sizes[1] / 2;    // 800000
  const int N = in_sizes[0] / NDIM; // 50000

  const int n4 = out_size / 4;
  k_zero<<<(n4 + 255) / 256, 256, 0, stream>>>((float4*)out, n4);
  k_prep<<<1, 256, 0, stream>>>(W1e, W2e, W1n, W2n, wts);
  k_prex<<<(n4 + 255) / 256, 256, 0, stream>>>((const float4*)x, (uint2*)xb, n4);
  k_edge<<<NB_EDGE, 128, 0, stream>>>(xb, eidx, eattr, wts, b1e, b2e, out, E);
  k_node<<<(N + 63) / 64, 128, 0, stream>>>(xb, x, wts, b1n, b2n, out, N);
}
